// HAGMoE_9646496546953
// MI455X (gfx1250) — matrix operand order repaired and checked by reading, not compiled, not run on hardware
//
#include <hip/hip_runtime.h>

// Problem sizes (fixed by the reference)
#define B_SZ   1024
#define H_SZ   768
#define I_SZ   3072
#define G_SZ   3
#define E_SZ   8
#define GE_SZ  24

typedef __attribute__((ext_vector_type(16))) __bf16 v16bf;
typedef __attribute__((ext_vector_type(8)))  float  v8f;

// float -> bf16 (round-to-nearest-even) as raw u16.
__device__ __forceinline__ unsigned short f2bf(float x) {
  unsigned u = __float_as_uint(x);
  return (unsigned short)((u + 0x7FFFu + ((u >> 16) & 1u)) >> 16);
}

// ---- async global->LDS copy helpers (CDNA5 ASYNCcnt path when available) --
#if defined(__HIP_DEVICE_COMPILE__) && \
    __has_builtin(__builtin_amdgcn_global_load_async_to_lds_b64) && \
    __has_builtin(__builtin_amdgcn_global_load_async_to_lds_b128) && \
    __has_builtin(__builtin_amdgcn_s_wait_asynccnt)
#define CDNA5_ASYNC_LDS 1
#endif

#ifdef CDNA5_ASYNC_LDS
typedef __attribute__((ext_vector_type(2))) int i32x2;
typedef __attribute__((ext_vector_type(4))) int i32x4;
typedef __attribute__((address_space(1))) i32x2* gp2_t;
typedef __attribute__((address_space(3))) i32x2* lp2_t;
typedef __attribute__((address_space(1))) i32x4* gp4_t;
typedef __attribute__((address_space(3))) i32x4* lp4_t;
__device__ __forceinline__ void cp_b64(unsigned short* d, const unsigned short* s) {
  __builtin_amdgcn_global_load_async_to_lds_b64((gp2_t)s, (lp2_t)d, 0, 0);
}
__device__ __forceinline__ void cp_b128(unsigned short* d, const unsigned short* s) {
  __builtin_amdgcn_global_load_async_to_lds_b128((gp4_t)s, (lp4_t)d, 0, 0);
}
__device__ __forceinline__ void cp_wait() { __builtin_amdgcn_s_wait_asynccnt(0); }
#else
__device__ __forceinline__ void cp_b64(unsigned short* d, const unsigned short* s) {
  *(uint2*)d = *(const uint2*)s;
}
__device__ __forceinline__ void cp_b128(unsigned short* d, const unsigned short* s) {
  *(uint4*)d = *(const uint4*)s;
}
__device__ __forceinline__ void cp_wait() {}
#endif

// --- WMMA fragment loaders (layouts per CDNA5 ISA 7.12.2, wave32) ---------
__device__ __forceinline__ v16bf load_a_frag(const unsigned short* base, int pitch) {
  int lane = threadIdx.x & 31;
  int m = lane & 15, kh = lane >> 4;
  const unsigned short* p = base + m * pitch + kh * 8;
  v16bf f;
  ((uint4*)&f)[0] = *(const uint4*)(p);
  ((uint4*)&f)[1] = *(const uint4*)(p + 16);
  return f;
}
__device__ __forceinline__ v16bf load_b_frag(const unsigned short* base, int pitch) {
  int lane = threadIdx.x & 31;
  int n = lane & 15, kh = lane >> 4;
  const unsigned short* p = base + n * pitch + kh * 8;
  v16bf f;
  ((uint4*)&f)[0] = *(const uint4*)(p);
  ((uint4*)&f)[1] = *(const uint4*)(p + 16);
  return f;
}

#define KP  40   // k-piece LDS pitch (bf16): 20 dwords -> conflict-free
#define BT  64
#define ICH 128
#define H1P 136  // 68 dwords -> conflict-free
#define XP  776  // full-X pitch: 388 dwords -> conflict-free

// ---------------- Kernel 0: fp32 -> bf16 activations ----------------------
__global__ void k_convert(const float* __restrict__ hf, const float* __restrict__ ha,
                          unsigned short* __restrict__ Xbf, unsigned short* __restrict__ Cat) {
  int i = blockIdx.x * blockDim.x + threadIdx.x;
  if (i >= B_SZ * H_SZ) return;
  int b = i / H_SZ, h = i % H_SZ;
  unsigned short xf = f2bf(hf[i]);
  Xbf[i] = xf;
  Cat[(size_t)b * (2 * H_SZ) + h] = xf;
  Cat[(size_t)b * (2 * H_SZ) + H_SZ + h] = f2bf(ha[i]);
}

// ---------------- residual init out = h_fused -----------------------------
__global__ void k_init_out(const float* __restrict__ hf, float* __restrict__ out) {
  int i = blockIdx.x * blockDim.x + threadIdx.x;
  if (i < B_SZ * H_SZ) out[i] = hf[i];
}

// ---------------- Kernel 1: h_cond = [hf|ha] @ Wc + bc (WMMA) -------------
__global__ __launch_bounds__(256) void k_cond_gemm(
    const unsigned short* __restrict__ Cat, const float* __restrict__ Wc,
    const float* __restrict__ bc, float* __restrict__ hcond) {
  __shared__ __align__(16) unsigned short As[64 * KP];
  __shared__ __align__(16) unsigned short Ws[128 * KP];
  int tid = threadIdx.x, wave = tid >> 5, lane = tid & 31;
  int b0 = blockIdx.x * 64, n0 = blockIdx.y * 128;
  int wr = wave >> 1, wc = wave & 1;
  int kh = lane >> 4, nl = lane & 15;
  v8f acc[4] = {};
  for (int kk = 0; kk < 2 * H_SZ; kk += 32) {
    __syncthreads();
    { int r = tid >> 2, s = tid & 3;
      *(uint4*)&As[r * KP + s * 8] =
          *(const uint4*)(Cat + (size_t)(b0 + r) * (2 * H_SZ) + kk + s * 8);
    }
    { int k = tid >> 3, cb = (tid & 7) * 16;
      const float* src = Wc + (size_t)(kk + k) * H_SZ + n0 + cb;
      for (int c = 0; c < 16; ++c) Ws[(cb + c) * KP + k] = f2bf(src[c]);
    }
    __syncthreads();
    v16bf a = load_a_frag(As + wr * 16 * KP, KP);
    for (int ct = 0; ct < 4; ++ct) {
      v16bf bfr = load_b_frag(Ws + (wc * 64 + ct * 16) * KP, KP);
      acc[ct] = __builtin_amdgcn_wmma_f32_16x16x32_bf16(
          false, a, false, bfr, (short)0, acc[ct], false, false);
    }
  }
  for (int ct = 0; ct < 4; ++ct) {
    int col = n0 + wc * 64 + ct * 16 + nl;
    float bias = bc[col];
    for (int r = 0; r < 8; ++r) {
      int row = b0 + wr * 16 + r + 8 * kh;
      hcond[(size_t)row * H_SZ + col] = acc[ct][r] + bias;
    }
  }
}

// ---------------- Kernel 2: routers -> combined weight w[b,g,e] -----------
__global__ void k_router(const float* __restrict__ hf, const float* __restrict__ hcond,
                         const float* __restrict__ Wg, const float* __restrict__ bg,
                         const float* __restrict__ Wr, const float* __restrict__ br,
                         float* __restrict__ wgt) {
  int b = blockIdx.x;
  int lane = threadIdx.x;
  float logits[27];
  for (int idx = 0; idx < 27; ++idx) {
    float s = 0.f;
    if (idx < 3) {
      for (int h = lane; h < H_SZ; h += 32)
        s += hf[(size_t)b * H_SZ + h] * Wg[(size_t)h * G_SZ + idx];
    } else {
      int j = idx - 3, g = j >> 3, e = j & 7;
      for (int h = lane; h < H_SZ; h += 32)
        s += hcond[(size_t)b * H_SZ + h] * Wr[((size_t)g * H_SZ + h) * E_SZ + e];
    }
    for (int o = 16; o > 0; o >>= 1) s += __shfl_xor(s, o, 32);
    s += (idx < 3) ? bg[idx] : br[idx - 3];
    logits[idx] = s;
  }
  if (lane == 0) {
    float m0 = fmaxf(logits[0], fmaxf(logits[1], logits[2]));
    float e0 = expf(logits[0] - m0), e1 = expf(logits[1] - m0), e2 = expf(logits[2] - m0);
    float sg = e0 + e1 + e2;
    float pg[3] = {e0 / sg, e1 / sg, e2 / sg};
    for (int g = 0; g < G_SZ; ++g) {
      float m = logits[3 + g * 8];
      for (int e = 1; e < 8; ++e) m = fmaxf(m, logits[3 + g * 8 + e]);
      float ex[8], se = 0.f;
      for (int e = 0; e < 8; ++e) { ex[e] = expf(logits[3 + g * 8 + e] - m); se += ex[e]; }
      for (int e = 0; e < 8; ++e)
        wgt[(size_t)b * GE_SZ + g * 8 + e] = pg[g] * ex[e] / se;
    }
  }
}

// ------- weight pre-tiling: fp32 -> bf16, laid out as the exact LDS image --
// W1 pieces: per (ge, chunk c, kpiece kkidx): [128 n][KP] (elems), flat.
__global__ __launch_bounds__(256) void k_tile_w1(const float* __restrict__ W1,
                                                 unsigned short* __restrict__ W1t) {
  __shared__ __align__(16) unsigned short T[ICH * KP];
  int blk = blockIdx.x;                       // ge*576 + c*24 + kkidx
  int ge = blk / 576, rem = blk % 576, c = rem / 24, kkidx = rem % 24;
  const float* src = W1 + ((size_t)ge * H_SZ + kkidx * 32) * I_SZ + c * ICH;
  int tid = threadIdx.x;
  { int k = tid >> 3, nb = (tid & 7) * 16;
    const float* s = src + (size_t)k * I_SZ + nb;
    for (int n = 0; n < 16; ++n) T[(nb + n) * KP + k] = f2bf(s[n]); }
  __syncthreads();
  uint2* d2 = (uint2*)(W1t + (size_t)blk * (ICH * KP));
  const uint2* t2 = (const uint2*)T;
  for (int i = tid; i < ICH * KP / 4; i += 256) d2[i] = t2[i];
}
// W2 pieces: per (ge, c, nh, kkidx): [384 n][KP] (elems), flat.
__global__ __launch_bounds__(256) void k_tile_w2(const float* __restrict__ W2,
                                                 unsigned short* __restrict__ W2t) {
  __shared__ __align__(16) unsigned short T[384 * KP];
  int blk = blockIdx.x;                       // ge*192 + c*8 + nh*4 + kkidx
  int ge = blk / 192, rem = blk % 192, c = rem / 8, r2 = rem % 8;
  int nh = r2 >> 2, kkidx = r2 & 3;
  const float* src = W2 + ((size_t)ge * I_SZ + c * ICH + kkidx * 32) * H_SZ + nh * 384;
  int tid = threadIdx.x;
  { int k = tid >> 3, nb = (tid & 7) * 48;
    const float* s = src + (size_t)k * H_SZ + nb;
    for (int n = 0; n < 48; ++n) T[(nb + n) * KP + k] = f2bf(s[n]); }
  __syncthreads();
  uint2* d2 = (uint2*)(W2t + (size_t)blk * (384 * KP));
  const uint2* t2 = (const uint2*)T;
  for (int i = tid; i < 384 * KP / 4; i += 256) d2[i] = t2[i];
}

// ---------------- fused dense MoE, fast path ------------------------------
// X tile (64x768 bf16, 97KB) staged once per block; weight pieces arrive as
// flat async global->LDS copies of the pre-tiled bf16 image (zero VALU).
__global__ __launch_bounds__(256) void k_moe_fast(
    const unsigned short* __restrict__ Xbf,
    const unsigned short* __restrict__ W1t, const float* __restrict__ b1,
    const unsigned short* __restrict__ W2t, const float* __restrict__ b2,
    const float* __restrict__ wgt, float* __restrict__ out) {
  __shared__ __align__(16) unsigned short Xs[BT * XP];    // 97.0 KB
  __shared__ __align__(16) unsigned short W1s[ICH * KP];  // 10.0 KB
  __shared__ __align__(16) unsigned short W2s[384 * KP];  // 30.0 KB
  __shared__ __align__(16) unsigned short H1s[BT * H1P];  // 17.0 KB
  __shared__ float wl[BT];
  int tid = threadIdx.x, wave = tid >> 5, lane = tid & 31;
  int b0 = blockIdx.x * BT;
  int ge = blockIdx.y;
  const unsigned short* W1e = W1t + (size_t)ge * 576 * (ICH * KP);
  const unsigned short* W2e = W2t + (size_t)ge * 192 * (384 * KP);
  const float* b1p = b1 + (size_t)ge * I_SZ;
  const float* b2p = b2 + (size_t)ge * H_SZ;
  if (tid < BT) wl[tid] = wgt[(size_t)(b0 + tid) * GE_SZ + ge];
  // whole X tile, once (6144 x b128)
  for (int i = 0; i < 24; ++i) {
    int idx = i * 256 + tid;
    int r = idx / 96, s = idx % 96;
    cp_b128(&Xs[r * XP + s * 8], Xbf + (size_t)(b0 + r) * H_SZ + s * 8);
  }
  int wr1 = wave >> 1, wc1 = wave & 1;
  int kh = lane >> 4, nl = lane & 15;
  v8f c2[2][4][3] = {};
  for (int c = 0; c < I_SZ / ICH; ++c) {
    v8f c1[4] = {};
    for (int kkidx = 0; kkidx < H_SZ / 32; ++kkidx) {
      __syncthreads();
      { const unsigned short* ps = W1e + ((size_t)c * 24 + kkidx) * (ICH * KP);
        for (int i = 0; i < 5; ++i) { int o = (i * 256 + tid) * 4; cp_b64(&W1s[o], ps + o); } }
      cp_wait();
      __syncthreads();
      v16bf a = load_a_frag(Xs + wr1 * 16 * XP + kkidx * 32, XP);
      for (int ct = 0; ct < 4; ++ct) {
        v16bf bfr = load_b_frag(W1s + (wc1 * 64 + ct * 16) * KP, KP);
        c1[ct] = __builtin_amdgcn_wmma_f32_16x16x32_bf16(
            false, a, false, bfr, (short)0, c1[ct], false, false);
      }
    }
    __syncthreads();
    for (int ct = 0; ct < 4; ++ct) {
      int col = wc1 * 64 + ct * 16 + nl;
      float bias = b1p[c * ICH + col];
      for (int r = 0; r < 8; ++r) {
        int row = wr1 * 16 + r + 8 * kh;
        float x = c1[ct][r] + bias;
        float gx = 0.5f * x * (1.0f + erff(x * 0.70710678118f));
        H1s[row * H1P + col] = f2bf(gx);
      }
    }
    for (int nh = 0; nh < 2; ++nh) {
      for (int kx = 0; kx < 4; ++kx) {
        __syncthreads();
        { const unsigned short* ps = W2e + (((size_t)c * 2 + nh) * 4 + kx) * (384 * KP);
          for (int i = 0; i < 15; ++i) { int o = (i * 256 + tid) * 4; cp_b64(&W2s[o], ps + o); } }
        cp_wait();
        __syncthreads();
        v16bf bfr[3];
        for (int ct = 0; ct < 3; ++ct)
          bfr[ct] = load_b_frag(W2s + (wave * 48 + ct * 16) * KP, KP);
        for (int rt = 0; rt < 4; ++rt) {
          v16bf a = load_a_frag(H1s + rt * 16 * H1P + kx * 32, H1P);
          for (int ct = 0; ct < 3; ++ct)
            c2[nh][rt][ct] = __builtin_amdgcn_wmma_f32_16x16x32_bf16(
                false, a, false, bfr[ct], (short)0, c2[nh][rt][ct], false, false);
        }
      }
    }
  }
  for (int nh = 0; nh < 2; ++nh)
    for (int rt = 0; rt < 4; ++rt)
      for (int ct = 0; ct < 3; ++ct) {
        int col = nh * 384 + wave * 48 + ct * 16 + nl;
        float bias = b2p[col];
        for (int r = 0; r < 8; ++r) {
          int rloc = rt * 16 + r + 8 * kh;
          float v = (c2[nh][rt][ct][r] + bias) * wl[rloc];
          atomicAdd(out + (size_t)(b0 + rloc) * H_SZ + col, v);
        }
      }
}

// ---------------- fused dense MoE, fallback (fp32 weights, small ws) ------
__global__ __launch_bounds__(256) void k_moe(
    const unsigned short* __restrict__ Xbf, const float* __restrict__ W1,
    const float* __restrict__ b1, const float* __restrict__ W2,
    const float* __restrict__ b2, const float* __restrict__ wgt,
    float* __restrict__ out) {
  __shared__ __align__(16) unsigned short Xs[BT * KP];
  __shared__ __align__(16) unsigned short W1s[ICH * KP];
  __shared__ __align__(16) unsigned short W2s[384 * KP];
  __shared__ __align__(16) unsigned short H1s[BT * H1P];
  __shared__ float wl[BT];
  int tid = threadIdx.x, wave = tid >> 5, lane = tid & 31;
  int b0 = blockIdx.x * BT;
  int ge = blockIdx.y;
  const float* W1p = W1 + (size_t)ge * H_SZ * I_SZ;
  const float* W2p = W2 + (size_t)ge * I_SZ * H_SZ;
  const float* b1p = b1 + (size_t)ge * I_SZ;
  const float* b2p = b2 + (size_t)ge * H_SZ;
  if (tid < BT) wl[tid] = wgt[(size_t)(b0 + tid) * GE_SZ + ge];
  int wr1 = wave >> 1, wc1 = wave & 1;
  int kh = lane >> 4, nl = lane & 15;
  v8f c2[2][4][3] = {};
  for (int i0 = 0; i0 < I_SZ; i0 += ICH) {
    v8f c1[4] = {};
    for (int kk = 0; kk < H_SZ; kk += 32) {
      __syncthreads();
      { int r = tid >> 2, s = tid & 3;
        *(uint4*)&Xs[r * KP + s * 8] =
            *(const uint4*)(Xbf + (size_t)(b0 + r) * H_SZ + kk + s * 8); }
      { int k = tid >> 3, cb = (tid & 7) * 16;
        const float* src = W1p + (size_t)(kk + k) * I_SZ + i0 + cb;
        for (int c = 0; c < 16; ++c) W1s[(cb + c) * KP + k] = f2bf(src[c]); }
      __syncthreads();
      v16bf a = load_a_frag(Xs + wr1 * 16 * KP, KP);
      for (int ct = 0; ct < 4; ++ct) {
        v16bf bfr = load_b_frag(W1s + (wc1 * 64 + ct * 16) * KP, KP);
        c1[ct] = __builtin_amdgcn_wmma_f32_16x16x32_bf16(
            false, a, false, bfr, (short)0, c1[ct], false, false);
      }
    }
    __syncthreads();
    for (int ct = 0; ct < 4; ++ct) {
      int col = wc1 * 64 + ct * 16 + nl;
      float bias = b1p[i0 + col];
      for (int r = 0; r < 8; ++r) {
        int row = wr1 * 16 + r + 8 * kh;
        float x = c1[ct][r] + bias;
        float gx = 0.5f * x * (1.0f + erff(x * 0.70710678118f));
        H1s[row * H1P + col] = f2bf(gx);
      }
    }
    for (int nh = 0; nh < 2; ++nh) {
      for (int kk = 0; kk < ICH; kk += 32) {
        __syncthreads();
        { int k = tid >> 3, cb = (tid & 7) * 48;
          const float* src = W2p + (size_t)(i0 + kk + k) * H_SZ + nh * 384 + cb;
          for (int c = 0; c < 48; ++c) W2s[(cb + c) * KP + k] = f2bf(src[c]); }
        __syncthreads();
        v16bf bfr[3];
        for (int ct = 0; ct < 3; ++ct)
          bfr[ct] = load_b_frag(W2s + (wave * 48 + ct * 16) * KP, KP);
        for (int rt = 0; rt < 4; ++rt) {
          v16bf a = load_a_frag(H1s + rt * 16 * H1P + kk, H1P);
          for (int ct = 0; ct < 3; ++ct)
            c2[nh][rt][ct] = __builtin_amdgcn_wmma_f32_16x16x32_bf16(
                false, a, false, bfr[ct], (short)0, c2[nh][rt][ct], false, false);
        }
      }
    }
  }
  for (int nh = 0; nh < 2; ++nh)
    for (int rt = 0; rt < 4; ++rt)
      for (int ct = 0; ct < 3; ++ct) {
        int col = nh * 384 + wave * 48 + ct * 16 + nl;
        float bias = b2p[col];
        for (int r = 0; r < 8; ++r) {
          int rloc = rt * 16 + r + 8 * kh;
          float v = (c2[nh][rt][ct][r] + bias) * wl[rloc];
          atomicAdd(out + (size_t)(b0 + rloc) * H_SZ + col, v);
        }
      }
}

// ---------------- host-side launcher --------------------------------------
extern "C" void kernel_launch(void* const* d_in, const int* in_sizes, int n_in,
                              void* d_out, int out_size, void* d_ws, size_t ws_size,
                              hipStream_t stream) {
  (void)in_sizes; (void)n_in; (void)out_size;
  const float* hf = (const float*)d_in[0];
  const float* ha = (const float*)d_in[1];
  const float* Wc = (const float*)d_in[2];
  const float* bc = (const float*)d_in[3];
  const float* Wg = (const float*)d_in[4];
  const float* bg = (const float*)d_in[5];
  const float* Wr = (const float*)d_in[6];
  const float* br = (const float*)d_in[7];
  const float* W1 = (const float*)d_in[8];
  const float* b1 = (const float*)d_in[9];
  const float* W2 = (const float*)d_in[10];
  const float* b2 = (const float*)d_in[11];
  float* out = (float*)d_out;

  char* ws = (char*)d_ws;
  unsigned short* Xbf   = (unsigned short*)(ws);                // B*H bf16
  unsigned short* Cat   = (unsigned short*)(ws + 1572864);      // B*2H bf16
  float*          hcond = (float*)(ws + 4718592);               // B*H f32
  float*          wgtb  = (float*)(ws + 7864320);               // B*24 f32
  const size_t W1T_OFF   = 7962624;
  const size_t W1T_BYTES = (size_t)24 * 576 * ICH * KP * 2;     // 141,557,760
  const size_t W2T_OFF   = W1T_OFF + W1T_BYTES;
  const size_t W2T_BYTES = (size_t)24 * 192 * 384 * KP * 2;     // 141,557,760
  const size_t FAST_NEED = W2T_OFF + W2T_BYTES;                 // ~278 MB

  k_convert<<<dim3((B_SZ * H_SZ + 255) / 256), dim3(256), 0, stream>>>(hf, ha, Xbf, Cat);
  k_cond_gemm<<<dim3(B_SZ / 64, H_SZ / 128), dim3(256), 0, stream>>>(Cat, Wc, bc, hcond);
  k_router<<<dim3(B_SZ), dim3(32), 0, stream>>>(hf, hcond, Wg, bg, Wr, br, wgtb);
  k_init_out<<<dim3((B_SZ * H_SZ + 255) / 256), dim3(256), 0, stream>>>(hf, out);

  if (ws_size >= FAST_NEED) {
    unsigned short* W1t = (unsigned short*)(ws + W1T_OFF);
    unsigned short* W2t = (unsigned short*)(ws + W2T_OFF);
    k_tile_w1<<<dim3(24 * 576), dim3(256), 0, stream>>>(W1, W1t);
    k_tile_w2<<<dim3(24 * 192), dim3(256), 0, stream>>>(W2, W2t);
    k_moe_fast<<<dim3(B_SZ / BT, GE_SZ), dim3(256), 0, stream>>>(Xbf, W1t, b1, W2t, b2, wgtb, out);
  } else {
    k_moe<<<dim3(B_SZ / BT, GE_SZ), dim3(256), 0, stream>>>(Xbf, W1, b1, W2, b2, wgtb, out);
  }
}
